// PVConv_1_58428735094798
// MI455X (gfx1250) — matrix operand order repaired and checked by reading, not compiled, not run on hardware
//
#include <hip/hip_runtime.h>
#include <hip/hip_bf16.h>

typedef __attribute__((ext_vector_type(16))) _Float16 v16h;
typedef __attribute__((ext_vector_type(8)))  float    v8f;

#define NPTS 8192
#define RG   32
#define NVOXG 32768   // 32^3

// ---------------------------------------------------------------------------
// Weight pack: w[co][ci][tap] (f32) -> wp[co][tap*CIN + ci] (f16), tap-major K
// ---------------------------------------------------------------------------
__global__ void pack_w(const float* __restrict__ w, _Float16* __restrict__ wp,
                       int CO, int CIN, int TAPS) {
    int idx = blockIdx.x * 256 + threadIdx.x;
    int tot = CO * CIN * TAPS;
    if (idx >= tot) return;
    int co  = idx / (CIN * TAPS);
    int r   = idx % (CIN * TAPS);
    int ci  = r / TAPS;
    int tap = r % TAPS;
    wp[(size_t)co * (TAPS * CIN) + tap * CIN + ci] = (_Float16)w[idx];
}

// ---------------------------------------------------------------------------
// Per-batch coord mean (3) + max centered norm -> cstat[b*4 + {0,1,2,3}]
// ---------------------------------------------------------------------------
__global__ void coord_stats(const float* __restrict__ coords, float* __restrict__ cstat) {
    int b = blockIdx.x;
    __shared__ float red[256];
    __shared__ float mean[3];
    for (int axis = 0; axis < 3; ++axis) {
        float s = 0.f;
        for (int n = threadIdx.x; n < NPTS; n += 256)
            s += coords[(b * 3 + axis) * NPTS + n];
        red[threadIdx.x] = s; __syncthreads();
        for (int o = 128; o > 0; o >>= 1) {
            if (threadIdx.x < o) red[threadIdx.x] += red[threadIdx.x + o];
            __syncthreads();
        }
        if (threadIdx.x == 0) mean[axis] = red[0] / (float)NPTS;
        __syncthreads();
    }
    float mx = 0.f;
    for (int n = threadIdx.x; n < NPTS; n += 256) {
        float dx = coords[(b * 3 + 0) * NPTS + n] - mean[0];
        float dy = coords[(b * 3 + 1) * NPTS + n] - mean[1];
        float dz = coords[(b * 3 + 2) * NPTS + n] - mean[2];
        mx = fmaxf(mx, sqrtf(dx * dx + dy * dy + dz * dz));
    }
    red[threadIdx.x] = mx; __syncthreads();
    for (int o = 128; o > 0; o >>= 1) {
        if (threadIdx.x < o) red[threadIdx.x] = fmaxf(red[threadIdx.x], red[threadIdx.x + o]);
        __syncthreads();
    }
    if (threadIdx.x == 0) {
        cstat[b * 4 + 0] = mean[0];
        cstat[b * 4 + 1] = mean[1];
        cstat[b * 4 + 2] = mean[2];
        cstat[b * 4 + 3] = red[0];
    }
}

// ---------------------------------------------------------------------------
// Voxelize scatter: compute nc (clipped, scaled to [0,31]) and atomically
// accumulate feature sums + counts. gsum layout [b][C][32768].
// ---------------------------------------------------------------------------
__global__ void vox_scatter(const float* __restrict__ features,
                            const float* __restrict__ coords,
                            const float* __restrict__ cstat,
                            float* __restrict__ nc,
                            float* __restrict__ gsum,
                            float* __restrict__ cnt, int C) {
    int tid = blockIdx.x * 256 + threadIdx.x;
    if (tid >= 4 * NPTS) return;
    int b = tid / NPTS, n = tid % NPTS;
    const float* cs = cstat + b * 4;
    float scale = 0.5f / cs[3];
    int vi[3];
    #pragma unroll
    for (int i = 0; i < 3; ++i) {
        float c = coords[(b * 3 + i) * NPTS + n] - cs[i];
        float v = (c * scale + 0.5f) * (float)RG;
        v = fminf(fmaxf(v, 0.f), (float)(RG - 1));
        nc[(b * 3 + i) * NPTS + n] = v;
        vi[i] = (int)rintf(v);   // RNE, matches jnp.round
    }
    int flat = (vi[0] * RG + vi[1]) * RG + vi[2];
    atomicAdd(&cnt[b * NVOXG + flat], 1.f);
    for (int c = 0; c < C; ++c)
        atomicAdd(&gsum[((size_t)(b * C + c)) * NVOXG + flat],
                  features[(size_t)(b * C + c) * NPTS + n]);
}

__global__ void vox_norm(float* __restrict__ grid, const float* __restrict__ cnt,
                         int C, int total) {
    for (int i = blockIdx.x * 256 + threadIdx.x; i < total; i += gridDim.x * 256) {
        int b = i / (C * NVOXG);
        int v = i % NVOXG;
        grid[i] = grid[i] / fmaxf(cnt[b * NVOXG + v], 1.f);
    }
}

__global__ void f2h(const float* __restrict__ in, _Float16* __restrict__ out, int n) {
    for (int i = blockIdx.x * 256 + threadIdx.x; i < n; i += gridDim.x * 256)
        out[i] = (_Float16)in[i];
}

// ---------------------------------------------------------------------------
// Implicit-GEMM 3D conv via WMMA f32_16x16x32_f16.
//   x  : f16 activations [B][CIN][D*H*32]
//   wp : f16 packed weights [CO][TAPS*CIN]  (K tap-major)
//   out: f32 [B][CO][D*H*32]  (pre-BN, bias added)
// One block per (b, d, h-pair): M = CO, N = 64 (two W-rows), K staged 64 at a
// time (always within one tap since CIN % 64 == 0). 8 waves tile M x N with
// 32x16 or 32x32 per wave -> 4..8 WMMAs per stage per wave with frag reuse.
// ---------------------------------------------------------------------------
template <int CIN, int CO, int TAPS>
__global__ __launch_bounds__(256) void conv_wmma(
        const _Float16* __restrict__ x, const _Float16* __restrict__ wp,
        const float* __restrict__ bias, float* __restrict__ out, int D, int H) {
    constexpr int KT      = TAPS * CIN;
    constexpr int WAVES_M = CO / 32;        // 2 (CO=64) or 4 (CO=128)
    constexpr int WAVES_N = 8 / WAVES_M;    // 4 or 2
    constexpr int NPW     = 64 / WAVES_N;   // 16 or 32 cols per wave
    constexpr int MTILES  = 2;              // 2 x 16 rows per wave
    constexpr int NTILES  = NPW / 16;       // 1 or 2
    constexpr int ASTR    = 36;             // A row stride in u32 (72 halfs, 144B)
    constexpr int BSTR    = 36;             // B row stride in u32

    __shared__ __align__(16) _Float16 At[CO * 72];   // [CO][64] halfs (+pad)
    __shared__ __align__(16) _Float16 Bt[64 * 72];   // [n=64][k=64] halfs (+pad)

    const int NVOX = D * H * 32;
    const int H2   = H >> 1;
    int blk = blockIdx.x;
    int b   = blk / (D * H2);
    int rem = blk % (D * H2);
    int d   = rem / H2;
    int h0  = (rem % H2) * 2;
    int pbase = (d * H + h0) * 32;

    int wave = threadIdx.x >> 5, lane = threadIdx.x & 31;
    int wm = wave % WAVES_M, wn = wave / WAVES_M;
    int row0 = wm * 32;
    int rrow = lane & 15, rhalf = lane >> 4;

    v8f acc[MTILES][NTILES] = {};

    const uint4* wp4 = (const uint4*)wp;

    for (int kk = 0; kk < KT; kk += 64) {
        int tap = (TAPS == 1) ? 0 : (kk / CIN);
        int ci0 = kk - tap * CIN;
        int od = 0, oh = 0, ow = 0;
        if (TAPS != 1) { od = tap / 9 - 1; oh = (tap / 3) % 3 - 1; ow = tap % 3 - 1; }

        __syncthreads();
        // --- stage A chunk: CO x 64 halfs via b128 ---
        {
            constexpr int AQ  = CO * 8;       // uint4 words in chunk
            constexpr int APT = AQ / 256;     // 4 (CO=128) or 2 (CO=64)
            uint4* At4 = (uint4*)At;
            #pragma unroll
            for (int i = 0; i < APT; ++i) {
                int idx = threadIdx.x + i * 256;
                int co = idx >> 3, q = idx & 7;
                At4[co * 9 + q] = wp4[(size_t)co * (KT / 8) + (kk >> 3) + q];
            }
        }
        // --- stage B chunk: 64 k x 64 n, transposed to [n][k] halfs ---
        {
            int k   = threadIdx.x >> 2;        // 0..63 -> ci = ci0 + k
            int n16 = (threadIdx.x & 3) << 4;  // 0,16,32,48
            const _Float16* src = x + ((size_t)(b * CIN + ci0 + k)) * NVOX;
            int dd = d + od;
            bool dok = (unsigned)dd < (unsigned)D;
            #pragma unroll
            for (int i = 0; i < 16; ++i) {
                int n = n16 + i;
                int r = n >> 5, w = n & 31;
                int hh = h0 + r + oh;
                int ww = w + ow;
                _Float16 v = (_Float16)0.f;
                if (dok && (unsigned)hh < (unsigned)H && (unsigned)ww < 32u)
                    v = src[(dd * H + hh) * 32 + ww];
                Bt[n * 72 + k] = v;
            }
        }
        // --- prefetch next weight stage (global_prefetch_b8) ---
        if (kk + 64 < KT)
            __builtin_prefetch((const char*)(wp4 + (size_t)(threadIdx.x >> 3) * (KT / 8)
                                             + ((kk + 64) >> 3)), 0, 3);
        __syncthreads();

        // --- compute: 2 K-subchunks of 32 ---
        #pragma unroll
        for (int ks = 0; ks < 64; ks += 32) {
            // A fragments (ISA 16-bit A layout), built from ds_read_b32
            v16h afr[MTILES];
            #pragma unroll
            for (int mt = 0; mt < MTILES; ++mt) {
                const unsigned* ap =
                    (const unsigned*)At + (row0 + mt * 16 + rrow) * ASTR + (ks >> 1);
                union { unsigned u[8]; v16h h; } cv;
                #pragma unroll
                for (int v = 0; v < 8; ++v)
                    cv.u[v] = ap[((v & 4) << 1) + (v & 3) + (rhalf << 2)];
                afr[mt] = cv.h;
            }
            #pragma unroll
            for (int nt = 0; nt < NTILES; ++nt) {
                const unsigned* bp =
                    (const unsigned*)Bt + (wn * NPW + nt * 16 + rrow) * BSTR + (ks >> 1);
                union { unsigned u[8]; v16h h; } cv;
                #pragma unroll
                for (int v = 0; v < 8; ++v)
                    cv.u[v] = bp[((v & 4) << 1) + (v & 3) + (rhalf << 2)];
                v16h bfr = cv.h;
                #pragma unroll
                for (int mt = 0; mt < MTILES; ++mt)
                    acc[mt][nt] = __builtin_amdgcn_wmma_f32_16x16x32_f16(
                        false, afr[mt], false, bfr, (short)0, acc[mt][nt], false, false);
            }
        }
    }

    // --- store: C/D 16x16 f32 layout: lane = N col, vgpr v -> M = v + 8*half ---
    int colN = lane & 15, half = lane >> 4;
    #pragma unroll
    for (int mt = 0; mt < MTILES; ++mt) {
        #pragma unroll
        for (int nt = 0; nt < NTILES; ++nt) {
            int n = wn * NPW + nt * 16 + colN;
            #pragma unroll
            for (int v = 0; v < 8; ++v) {
                int co = row0 + mt * 16 + v + (half << 3);
                out[((size_t)(b * CO + co)) * NVOX + pbase + n] = acc[mt][nt][v] + bias[co];
            }
        }
    }
}

// ---------------------------------------------------------------------------
// Per-channel batch stats over [B][C][nvox] -> mean[c], biased var[c]
// ---------------------------------------------------------------------------
__global__ void ch_stats(const float* __restrict__ x, float* __restrict__ mean,
                         float* __restrict__ var, int C, int nvox, int nb) {
    int c = blockIdx.x;
    __shared__ float rs[256], rq[256];
    float s = 0.f, q = 0.f;
    int total = nb * nvox;
    for (int i = threadIdx.x; i < total; i += 256) {
        int b = i / nvox, v = i % nvox;
        float val = x[((size_t)(b * C + c)) * nvox + v];
        s += val; q += val * val;
    }
    rs[threadIdx.x] = s; rq[threadIdx.x] = q; __syncthreads();
    for (int o = 128; o > 0; o >>= 1) {
        if (threadIdx.x < o) { rs[threadIdx.x] += rs[threadIdx.x + o]; rq[threadIdx.x] += rq[threadIdx.x + o]; }
        __syncthreads();
    }
    if (threadIdx.x == 0) {
        float m = rs[0] / (float)total;
        mean[c] = m;
        var[c]  = rq[0] / (float)total - m * m;
    }
}

// ---------------------------------------------------------------------------
// Fused BN apply (+skip)(+relu), optional f32 and/or f16 outputs
// ---------------------------------------------------------------------------
__global__ void bn_apply(const float* __restrict__ x, const float* __restrict__ mean,
                         const float* __restrict__ var, const float* __restrict__ g,
                         const float* __restrict__ be, const float* __restrict__ skip,
                         float* __restrict__ outf, _Float16* __restrict__ outh,
                         int C, int nvox, int total, int relu) {
    for (int i = blockIdx.x * 256 + threadIdx.x; i < total; i += gridDim.x * 256) {
        int c = (i / nvox) % C;
        float y = (x[i] - mean[c]) * rsqrtf(var[c] + 1e-5f) * g[c] + be[c];
        if (skip) y += skip[i];
        if (relu) y = fmaxf(y, 0.f);
        if (outf) outf[i] = y;
        if (outh) outh[i] = (_Float16)y;
    }
}

// ---------------------------------------------------------------------------
// Devoxelize (trilinear gather) and accumulate into out (which holds pt branch)
// ---------------------------------------------------------------------------
__global__ void devox_add(const float* __restrict__ grid, const float* __restrict__ nc,
                          float* __restrict__ out, int C) {
    int tid = blockIdx.x * 256 + threadIdx.x;
    int total = 4 * C * NPTS;
    if (tid >= total) return;
    int n = tid % NPTS;
    int c = (tid / NPTS) % C;
    int b = tid / (C * NPTS);
    float x = nc[(b * 3 + 0) * NPTS + n];
    float y = nc[(b * 3 + 1) * NPTS + n];
    float z = nc[(b * 3 + 2) * NPTS + n];
    float fx = fminf(floorf(x), 31.f), fy = fminf(floorf(y), 31.f), fz = fminf(floorf(z), 31.f);
    int x0 = (int)fx, y0 = (int)fy, z0 = (int)fz;
    float wx = x - fx, wy = y - fy, wz = z - fz;
    const float* g = grid + ((size_t)(b * C + c)) * NVOXG;
    float acc = 0.f;
    #pragma unroll
    for (int dx = 0; dx < 2; ++dx) {
        int ix = min(x0 + dx, 31);
        float wxx = dx ? wx : 1.f - wx;
        #pragma unroll
        for (int dy = 0; dy < 2; ++dy) {
            int iy = min(y0 + dy, 31);
            float wyy = dy ? wy : 1.f - wy;
            #pragma unroll
            for (int dz = 0; dz < 2; ++dz) {
                int iz = min(z0 + dz, 31);
                float wzz = dz ? wz : 1.f - wz;
                acc += g[(ix * RG + iy) * RG + iz] * (wxx * wyy * wzz);
            }
        }
    }
    out[tid] += acc;
}

// ---------------------------------------------------------------------------
extern "C" void kernel_launch(void* const* d_in, const int* in_sizes, int n_in,
                              void* d_out, int out_size, void* d_ws, size_t ws_size,
                              hipStream_t stream) {
    const float* features = (const float*)d_in[0];
    const float* coords   = (const float*)d_in[1];
    const float* b0_w  = (const float*)d_in[2];
    const float* b0_b  = (const float*)d_in[3];
    const float* b0_g  = (const float*)d_in[4];
    const float* b0_be = (const float*)d_in[5];
    const float* r1_w1 = (const float*)d_in[6];
    const float* r1_b1 = (const float*)d_in[7];
    const float* r1_g1 = (const float*)d_in[8];
    const float* r1_be1= (const float*)d_in[9];
    const float* r1_w2 = (const float*)d_in[10];
    const float* r1_b2 = (const float*)d_in[11];
    const float* r1_g2 = (const float*)d_in[12];
    const float* r1_be2= (const float*)d_in[13];
    const float* r1_wd = (const float*)d_in[14];
    const float* r1_bd = (const float*)d_in[15];
    const float* r1_gd = (const float*)d_in[16];
    const float* r1_bed= (const float*)d_in[17];
    const float* r2_w1 = (const float*)d_in[18];
    const float* r2_b1 = (const float*)d_in[19];
    const float* r2_g1 = (const float*)d_in[20];
    const float* r2_be1= (const float*)d_in[21];
    const float* r2_w2 = (const float*)d_in[22];
    const float* r2_b2 = (const float*)d_in[23];
    const float* r2_g2 = (const float*)d_in[24];
    const float* r2_be2= (const float*)d_in[25];
    const float* p_w   = (const float*)d_in[26];
    const float* p_b   = (const float*)d_in[27];
    const float* p_g   = (const float*)d_in[28];
    const float* p_be  = (const float*)d_in[29];

    // ---- workspace carve-up ----
    const size_t BIG = (size_t)4 * 128 * NVOXG;      // 16,777,216 elems
    char* p = (char*)d_ws;
    float*    X   = (float*)p;      p += BIG * 4;
    float*    Y   = (float*)p;      p += BIG * 4;
    float*    Z   = (float*)p;      p += BIG * 4;
    _Float16* Ah  = (_Float16*)p;   p += BIG * 2;
    _Float16* Bh  = (_Float16*)p;   p += BIG * 2;
    float*    cnt = (float*)p;      p += (size_t)4 * NVOXG * 4;
    float*    ncb = (float*)p;      p += (size_t)4 * 3 * NPTS * 4;
    float*    cstat = (float*)p;    p += 16 * 4;
    float*    meanb = (float*)p;    p += 128 * 4;
    float*    varb  = (float*)p;    p += 128 * 4;
    _Float16* w_b0   = (_Float16*)p; p += (size_t)64  * 1728 * 2;
    _Float16* w_r1w1 = (_Float16*)p; p += (size_t)128 * 1728 * 2;
    _Float16* w_r1wd = (_Float16*)p; p += (size_t)128 * 64   * 2;
    _Float16* w_r1w2 = (_Float16*)p; p += (size_t)128 * 3456 * 2;
    _Float16* w_r2w1 = (_Float16*)p; p += (size_t)128 * 3456 * 2;
    _Float16* w_r2w2 = (_Float16*)p; p += (size_t)128 * 3456 * 2;
    _Float16* w_pw   = (_Float16*)p; p += (size_t)128 * 64   * 2;

    float* out = (float*)d_out;
    auto cdiv = [](int a, int b) { return (a + b - 1) / b; };

    // ---- pack weights to f16 tap-major ----
    pack_w<<<cdiv(64 * 64 * 27, 256), 256, 0, stream>>>(b0_w, w_b0, 64, 64, 27);
    pack_w<<<cdiv(128 * 64 * 27, 256), 256, 0, stream>>>(r1_w1, w_r1w1, 128, 64, 27);
    pack_w<<<cdiv(128 * 64, 256), 256, 0, stream>>>(r1_wd, w_r1wd, 128, 64, 1);
    pack_w<<<cdiv(128 * 128 * 27, 256), 256, 0, stream>>>(r1_w2, w_r1w2, 128, 128, 27);
    pack_w<<<cdiv(128 * 128 * 27, 256), 256, 0, stream>>>(r2_w1, w_r2w1, 128, 128, 27);
    pack_w<<<cdiv(128 * 128 * 27, 256), 256, 0, stream>>>(r2_w2, w_r2w2, 128, 128, 27);
    pack_w<<<cdiv(128 * 64, 256), 256, 0, stream>>>(p_w, w_pw, 128, 64, 1);

    // ---- voxelize ----
    hipMemsetAsync(Y, 0, (size_t)4 * 64 * NVOXG * 4, stream);
    hipMemsetAsync(cnt, 0, (size_t)4 * NVOXG * 4, stream);
    coord_stats<<<4, 256, 0, stream>>>(coords, cstat);
    vox_scatter<<<cdiv(4 * NPTS, 256), 256, 0, stream>>>(features, coords, cstat, ncb, Y, cnt, 64);
    {
        int tot = 4 * 64 * NVOXG;
        vox_norm<<<cdiv(tot, 256), 256, 0, stream>>>(Y, cnt, 64, tot);
        f2h<<<cdiv(tot, 256), 256, 0, stream>>>(Y, Ah, tot);
    }

    const int GRID_VOX = 4 * 32 * 16;   // (b, d, h-pair) blocks
    const int T64  = 4 * 64  * NVOXG;
    const int T128 = 4 * 128 * NVOXG;

    // ---- b0: conv 64->64 + BN + ReLU  (v0 -> Y f32, Ah f16) ----
    conv_wmma<64, 64, 27><<<GRID_VOX, 256, 0, stream>>>(Ah, w_b0, b0_b, X, 32, 32);
    ch_stats<<<64, 256, 0, stream>>>(X, meanb, varb, 64, NVOXG, 4);
    bn_apply<<<cdiv(T64, 256), 256, 0, stream>>>(X, meanb, varb, b0_g, b0_be,
                                                 nullptr, Y, Ah, 64, NVOXG, T64, 1);

    // ---- r1 downsample: 1x1x1 conv 64->128 + BN (sc -> Z) ----
    conv_wmma<64, 128, 1><<<GRID_VOX, 256, 0, stream>>>(Ah, w_r1wd, r1_bd, X, 32, 32);
    ch_stats<<<128, 256, 0, stream>>>(X, meanb, varb, 128, NVOXG, 4);
    bn_apply<<<cdiv(T128, 256), 256, 0, stream>>>(X, meanb, varb, r1_gd, r1_bed,
                                                  nullptr, Z, nullptr, 128, NVOXG, T128, 0);

    // ---- r1 conv1: 64->128 + BN + ReLU (h1 -> Bh f16) ----
    conv_wmma<64, 128, 27><<<GRID_VOX, 256, 0, stream>>>(Ah, w_r1w1, r1_b1, X, 32, 32);
    ch_stats<<<128, 256, 0, stream>>>(X, meanb, varb, 128, NVOXG, 4);
    bn_apply<<<cdiv(T128, 256), 256, 0, stream>>>(X, meanb, varb, r1_g1, r1_be1,
                                                  nullptr, nullptr, Bh, 128, NVOXG, T128, 1);

    // ---- r1 conv2: 128->128 + BN + skip(Z) + ReLU (v1 -> Y f32, Ah f16) ----
    conv_wmma<128, 128, 27><<<GRID_VOX, 256, 0, stream>>>(Bh, w_r1w2, r1_b2, X, 32, 32);
    ch_stats<<<128, 256, 0, stream>>>(X, meanb, varb, 128, NVOXG, 4);
    bn_apply<<<cdiv(T128, 256), 256, 0, stream>>>(X, meanb, varb, r1_g2, r1_be2,
                                                  Z, Y, Ah, 128, NVOXG, T128, 1);

    // ---- r2 conv1: 128->128 + BN + ReLU (-> Bh f16) ----
    conv_wmma<128, 128, 27><<<GRID_VOX, 256, 0, stream>>>(Ah, w_r2w1, r2_b1, X, 32, 32);
    ch_stats<<<128, 256, 0, stream>>>(X, meanb, varb, 128, NVOXG, 4);
    bn_apply<<<cdiv(T128, 256), 256, 0, stream>>>(X, meanb, varb, r2_g1, r2_be1,
                                                  nullptr, nullptr, Bh, 128, NVOXG, T128, 1);

    // ---- r2 conv2: 128->128 + BN + skip(Y=v1) + ReLU (v2 -> Z f32) ----
    conv_wmma<128, 128, 27><<<GRID_VOX, 256, 0, stream>>>(Bh, w_r2w2, r2_b2, X, 32, 32);
    ch_stats<<<128, 256, 0, stream>>>(X, meanb, varb, 128, NVOXG, 4);
    bn_apply<<<cdiv(T128, 256), 256, 0, stream>>>(X, meanb, varb, r2_g2, r2_be2,
                                                  Y, Z, nullptr, 128, NVOXG, T128, 1);

    // ---- point branch: pt = BN(ReLU)(p_w @ features + p_b) -> d_out ----
    {
        int tf = 4 * 64 * NPTS;
        f2h<<<cdiv(tf, 256), 256, 0, stream>>>(features, Ah, tf);
        // treat 8192 points as D=8, H=32, W=32 "voxels"; 1x1x1 conv => pure GEMM
        conv_wmma<64, 128, 1><<<4 * 8 * 16, 256, 0, stream>>>(Ah, w_pw, p_b, X, 8, 32);
        int tp = 4 * 128 * NPTS;
        ch_stats<<<128, 256, 0, stream>>>(X, meanb, varb, 128, NPTS, 4);
        bn_apply<<<cdiv(tp, 256), 256, 0, stream>>>(X, meanb, varb, p_g, p_be,
                                                    nullptr, out, nullptr, 128, NPTS, tp, 1);
    }

    // ---- devoxelize(v2) accumulated onto pt in d_out ----
    devox_add<<<cdiv(4 * 128 * NPTS, 256), 256, 0, stream>>>(Z, ncb, out, 128);

    // ---- coords passthrough (second tuple output) ----
    hipMemcpyAsync(out + (size_t)4 * 128 * NPTS, coords,
                   (size_t)4 * 3 * NPTS * sizeof(float),
                   hipMemcpyDeviceToDevice, stream);
}
